// StratsOriginal_26182120636783
// MI455X (gfx1250) — matrix operand order repaired and checked by reading, not compiled, not run on hardware
//
#include <hip/hip_runtime.h>
#include <hip/hip_bf16.h>
#include <math.h>

// ---------------------------------------------------------------------------
// Model constants (from reference): B=16, L=2048, D=128, H=4, DK=32, N=4,
// DFF=256, F=129, DM=16, INT=11, V=130
// ---------------------------------------------------------------------------
#define BSZ   16
#define LSEQ  2048
#define DMODEL 128
#define NHEAD 4
#define DKH   32
#define NLAYER 4
#define DFF   256
#define FOUT  129
#define DMOG  16
#define INTH  11
#define FMINV (-3.402823466e38f)

typedef __attribute__((ext_vector_type(16))) __bf16 v16bf;
typedef __attribute__((ext_vector_type(8)))  __bf16 v8bf;
typedef __attribute__((ext_vector_type(8)))  float  v8f;

// fp32 -> bf16 bits, round-to-nearest-even, NaN safe, branchless (v_cndmask)
__device__ __forceinline__ unsigned short f2bf_bits(float f) {
  union { float f; unsigned u; } a; a.f = f;
  unsigned u = a.u;
  unsigned rne = (u + 0x7fffu + ((u >> 16) & 1u)) >> 16;
  unsigned nanv = (u >> 16) | 0x0040u;
  bool isnan = (u & 0x7fffffffu) > 0x7f800000u;
  return (unsigned short)(isnan ? nanv : rne);
}
// fast path: caller guarantees non-NaN input (e.g. softmax probabilities)
__device__ __forceinline__ __bf16 f2bf_fast(float f) {
  union { float f; unsigned u; } a; a.f = f;
  unsigned short h = (unsigned short)((a.u + 0x7fffu + ((a.u >> 16) & 1u)) >> 16);
  __bf16 b; __builtin_memcpy(&b, &h, 2); return b;
}

__device__ __forceinline__ v8f wmma_bf16(v16bf a, v16bf b, v8f c) {
  // D = A(16x32 bf16) * B(32x16 bf16) + C(16x16 f32)
  return __builtin_amdgcn_wmma_f32_16x16x32_bf16(false, a, false, b, (short)0, c,
                                                 false, false);
}

// A-operand loader: row-major bf16 tile, 16 rows x 32 K, `base` at tile origin,
// `stride` in elements. Uses contiguous K order (same as load_B):
//   lanes 0-15 : elems 0..15 -> K=0..15
//   lanes 16-31: elems 0..15 -> K=16..31
__device__ __forceinline__ v16bf load_A(const unsigned short* base, int stride, int lane) {
  int m = lane & 15, hi = lane >> 4;
  const unsigned short* p = base + (size_t)m * stride;
  v8bf lo = *reinterpret_cast<const v8bf*>(p + 16 * hi);
  v8bf h8 = *reinterpret_cast<const v8bf*>(p + 16 * hi + 8);
  v16bf r;
#pragma unroll
  for (int i = 0; i < 8; ++i) { r[i] = lo[i]; r[i + 8] = h8[i]; }
  return r;
}

// B-operand loader: tile stored transposed, i.e. memory is [N][K] row-major
// (`base` at tile origin, stride = K-row stride in elements).
//   lane -> N = lane&15 ; elems e -> K = (lane>>4)*16 + e   (contiguous 32B)
__device__ __forceinline__ v16bf load_B(const unsigned short* base, int stride, int lane) {
  int n = lane & 15, hi = lane >> 4;
  return *reinterpret_cast<const v16bf*>(base + (size_t)n * stride + 16 * hi);
}

// --- gfx1250 async global->LDS copy (ASYNCcnt tracked) ----------------------
__device__ __forceinline__ void async_copy16(unsigned lds_off, const void* gaddr) {
  // GLOBAL_LOAD_ASYNC_TO_LDS_B128: VDST = per-lane LDS byte address,
  // VADDR = 64-bit global address (GV mode, saddr = off/null).
  asm volatile("global_load_async_to_lds_b128 %0, %1, off"
               :: "v"(lds_off), "v"(gaddr) : "memory");
}
__device__ __forceinline__ void wait_async_le8() {
  asm volatile("s_wait_asynccnt 0x8" ::: "memory");
}
__device__ __forceinline__ void wait_async_le0() {
  asm volatile("s_wait_asynccnt 0x0" ::: "memory");
}

// ---------------------------------------------------------------------------
// Weight convert + transpose: src f32 [layers][rows][cols] -> dst bf16 [layers][cols][rows]
// ---------------------------------------------------------------------------
__global__ void k_convT(const float* __restrict__ src, unsigned short* __restrict__ dst,
                        int layers, int rows, int cols) {
  int idx = blockIdx.x * blockDim.x + threadIdx.x;
  int total = layers * rows * cols;
  if (idx >= total) return;
  int i = idx / (rows * cols);
  int rem = idx - i * rows * cols;
  int r = rem / cols, c = rem - r * cols;
  dst[((size_t)i * cols + c) * rows + r] = f2bf_bits(src[idx]);
}

// ---------------------------------------------------------------------------
// Embedding: x = CVE(time) + CVE(value) + var_table[var]   (block = one (b,l))
// ---------------------------------------------------------------------------
__global__ void k_embed(const float* __restrict__ values, const float* __restrict__ times,
                        const int* __restrict__ vars,
                        const float* __restrict__ W1t, const float* __restrict__ b1t,
                        const float* __restrict__ W2t,
                        const float* __restrict__ W1v, const float* __restrict__ b1v,
                        const float* __restrict__ W2v,
                        const float* __restrict__ var_table,
                        float* __restrict__ xf, unsigned short* __restrict__ xbf) {
  int bl = blockIdx.x;      // 0..B*L-1
  int d = threadIdx.x;      // 0..127
  __shared__ float ht[INTH], hv[INTH];
  if (d < INTH) {
    ht[d] = tanhf(times[bl] * W1t[d] + b1t[d]);
    hv[d] = tanhf(values[bl] * W1v[d] + b1v[d]);
  }
  __syncthreads();
  float acc = var_table[(size_t)vars[bl] * DMODEL + d];
#pragma unroll
  for (int j = 0; j < INTH; ++j)
    acc += ht[j] * W2t[j * DMODEL + d] + hv[j] * W2v[j * DMODEL + d];
  size_t idx = (size_t)bl * DMODEL + d;
  xf[idx] = acc;
  xbf[idx] = f2bf_bits(acc);
}

// ---------------------------------------------------------------------------
// Demographics MLP (block = one batch row, 256 threads)
// ---------------------------------------------------------------------------
__global__ void k_demo(const float* __restrict__ dm, const float* __restrict__ Wd1,
                       const float* __restrict__ bd1, const float* __restrict__ Wd2,
                       const float* __restrict__ bd2, float* __restrict__ demo) {
  int b = blockIdx.x, t = threadIdx.x;
  __shared__ float h[2 * DMODEL];
  float a = bd1[t];
#pragma unroll
  for (int k2 = 0; k2 < DMOG; ++k2) a += dm[b * DMOG + k2] * Wd1[k2 * 2 * DMODEL + t];
  h[t] = tanhf(a);
  __syncthreads();
  if (t < DMODEL) {
    float a2 = bd2[t];
    for (int j = 0; j < 2 * DMODEL; ++j) a2 += h[j] * Wd2[j * DMODEL + t];
    demo[b * DMODEL + t] = a2;
  }
}

// ---------------------------------------------------------------------------
// QKV projection: x[B*L,128] @ W^T -> q,k [B,H,L,32] bf16 ; v stored transposed
// [B,H,32,L].  One wave per 16x16 tile; jobs = 3 mats * 8 ntiles.
// ---------------------------------------------------------------------------
__global__ void k_qkv(const unsigned short* __restrict__ xbf,
                      const unsigned short* __restrict__ WqT,
                      const unsigned short* __restrict__ WkT,
                      const unsigned short* __restrict__ WvT,
                      unsigned short* __restrict__ q, unsigned short* __restrict__ k,
                      unsigned short* __restrict__ vT, int layer) {
  int wid = blockIdx.x * (blockDim.x >> 5) + (threadIdx.x >> 5);
  int lane = threadIdx.x & 31;
  int mtile = wid / 24;
  int job = wid % 24;
  int mat = job >> 3;       // 0=q 1=k 2=v
  int ntile = job & 7;      // over H*DK/16 = 8
  const unsigned short* WT = (mat == 0 ? WqT : (mat == 1 ? WkT : WvT));
  const unsigned short* Arow = xbf + (size_t)mtile * 16 * DMODEL;
  const unsigned short* Brow = WT + (size_t)layer * DMODEL * DMODEL + (size_t)ntile * 16 * DMODEL;
  v8f c = {};
#pragma unroll
  for (int kc = 0; kc < 4; ++kc)
    c = wmma_bf16(load_A(Arow + kc * 32, DMODEL, lane),
                  load_B(Brow + kc * 32, DMODEL, lane), c);
  int n = lane & 15, hi = lane >> 4;
  int hh = ntile >> 1, sub = ntile & 1;
#pragma unroll
  for (int r = 0; r < 8; ++r) {
    int M = r + 8 * hi;
    int lg = mtile * 16 + M;
    int bi = lg >> 11;            // / LSEQ
    int l = lg & (LSEQ - 1);
    unsigned short val = f2bf_bits(c[r]);
    int bh = bi * NHEAD + hh;
    if (mat == 2)
      vT[((size_t)bh * DKH + sub * 16 + n) * LSEQ + l] = val;
    else {
      unsigned short* dst = (mat == 0 ? q : k);
      dst[((size_t)bh * LSEQ + l) * DKH + sub * 16 + n] = val;
    }
  }
}

// ---------------------------------------------------------------------------
// Flash-style attention: one wave per (b,h, 32-query-row tile).
// K/V chunks (32 keys) double-buffered in LDS via async global->LDS loads.
// 8 WMMAs per chunk.  Softmax uses all 32 lanes (row = lane).
// No 1/sqrt(dk) scaling (matches reference).
// ---------------------------------------------------------------------------
__global__ void __launch_bounds__(32) k_attn(
    const unsigned short* __restrict__ q, const unsigned short* __restrict__ k,
    const unsigned short* __restrict__ vT, const unsigned char* __restrict__ mask,
    unsigned short* __restrict__ ao) {
  int wid = blockIdx.x;
  int mtile = wid & 63;                // 32-row tile within (b,h)
  int bh = wid >> 6;                   // 0..B*H-1
  int b = bh >> 2;
  int lane = threadIdx.x;
  int n = lane & 15, hi = lane >> 4;

  __shared__ __align__(64) unsigned short Kbuf[2][32 * 32];  // [key][kk]
  __shared__ __align__(64) unsigned short Vbuf[2][32 * 32];  // [e][key]
  __shared__ float S[32][33];
  __shared__ float rm[32], rl[32], rsc[32];
  __shared__ unsigned char qm[32];
  __shared__ unsigned char km[32];

  rm[lane] = -INFINITY; rl[lane] = 0.f;
  qm[lane] = mask[(size_t)b * LSEQ + mtile * 32 + lane];
  __syncthreads();

  const unsigned short* Qbase = q + ((size_t)bh * LSEQ + mtile * 32) * DKH;
  v16bf A0 = load_A(Qbase, DKH, lane);
  v16bf A1 = load_A(Qbase + 16 * DKH, DKH, lane);
  v8f O00 = {}, O01 = {}, O10 = {}, O11 = {};

  const unsigned short* Kg = k + (size_t)bh * LSEQ * DKH;    // chunk stride 2048B
  const unsigned short* Vg = vT + (size_t)bh * DKH * LSEQ;   // row stride 4096B
  unsigned klds[2] = { (unsigned)(size_t)&Kbuf[0][0], (unsigned)(size_t)&Kbuf[1][0] };
  unsigned vlds[2] = { (unsigned)(size_t)&Vbuf[0][0], (unsigned)(size_t)&Vbuf[1][0] };

  auto issue = [&](int c) {                      // 8 async b128 ops per stage
    int buf = c & 1;
    unsigned kl = klds[buf] + (unsigned)lane * 16u;
    const char* kg = (const char*)Kg + (size_t)c * 2048 + lane * 16;
#pragma unroll
    for (int j = 0; j < 4; ++j)
      async_copy16(kl + (unsigned)j * 512u, kg + j * 512);
    unsigned vlb = vlds[buf];
#pragma unroll
    for (int j = 0; j < 4; ++j) {
      int idx = j * 32 + lane;
      int row = idx >> 2, seg = idx & 3;         // 32 rows x 64B
      async_copy16(vlb + (unsigned)(row * 64 + seg * 16),
                   (const char*)Vg + (size_t)row * (LSEQ * 2) + (size_t)c * 64 + seg * 16);
    }
  };

  issue(0);
  for (int c = 0; c < LSEQ / 32; ++c) {
    if (c + 1 < LSEQ / 32) { issue(c + 1); wait_async_le8(); }  // stage c done
    else                   { wait_async_le0(); }
    int buf = c & 1;
    km[lane] = mask[(size_t)b * LSEQ + c * 32 + lane];
    const unsigned short* Kb = &Kbuf[buf][0];
    v16bf B0 = *reinterpret_cast<const v16bf*>(Kb + (size_t)n * DKH + 16 * hi);
    v16bf B1 = *reinterpret_cast<const v16bf*>(Kb + (size_t)(16 + n) * DKH + 16 * hi);
    v8f s00 = {}, s01 = {}, s10 = {}, s11 = {};
    s00 = wmma_bf16(A0, B0, s00);
    s01 = wmma_bf16(A0, B1, s01);
    s10 = wmma_bf16(A1, B0, s10);
    s11 = wmma_bf16(A1, B1, s11);
    __syncthreads();                  // prev-iter P reads done; km visible
#pragma unroll
    for (int r = 0; r < 8; ++r) {
      int M = r + 8 * hi, M2 = 16 + M;
      S[M][n]       = s00[r] + ((qm[M]  && km[n])      ? 0.f : FMINV);
      S[M][16 + n]  = s01[r] + ((qm[M]  && km[16 + n]) ? 0.f : FMINV);
      S[M2][n]      = s10[r] + ((qm[M2] && km[n])      ? 0.f : FMINV);
      S[M2][16 + n] = s11[r] + ((qm[M2] && km[16 + n]) ? 0.f : FMINV);
    }
    __syncthreads();
    {                                 // online softmax, row = lane (all 32 busy)
      float mo = rm[lane], cmax = -INFINITY;
#pragma unroll
      for (int j = 0; j < 32; ++j) cmax = fmaxf(cmax, S[lane][j]);
      float nm = fmaxf(mo, cmax);
      float sc = __expf(mo - nm);
      float sum = 0.f;
#pragma unroll
      for (int j = 0; j < 32; ++j) {
        float p = __expf(S[lane][j] - nm);
        S[lane][j] = p; sum += p;
      }
      rm[lane] = nm; rsc[lane] = sc; rl[lane] = rl[lane] * sc + sum;
    }
    __syncthreads();
#pragma unroll
    for (int r = 0; r < 8; ++r) {
      float sc0 = rsc[r + 8 * hi], sc1 = rsc[16 + r + 8 * hi];
      O00[r] *= sc0; O01[r] *= sc0;
      O10[r] *= sc1; O11[r] *= sc1;
    }
    v16bf P0, P1;                     // probabilities are non-NaN: fast cvt
#pragma unroll
    for (int e = 0; e < 8; ++e) {
      P0[e]     = f2bf_fast(S[n][16 * hi + e]);
      P0[e + 8] = f2bf_fast(S[n][16 * hi + 8 + e]);
      P1[e]     = f2bf_fast(S[16 + n][16 * hi + e]);
      P1[e + 8] = f2bf_fast(S[16 + n][16 * hi + 8 + e]);
    }
    const unsigned short* Vb = &Vbuf[buf][0];
    v16bf V0 = *reinterpret_cast<const v16bf*>(Vb + (size_t)n * 32 + 16 * hi);
    v16bf V1 = *reinterpret_cast<const v16bf*>(Vb + (size_t)(16 + n) * 32 + 16 * hi);
    O00 = wmma_bf16(P0, V0, O00);
    O01 = wmma_bf16(P0, V1, O01);
    O10 = wmma_bf16(P1, V0, O10);
    O11 = wmma_bf16(P1, V1, O11);
  }
  __syncthreads();
  int hh = bh & 3;
#pragma unroll
  for (int r = 0; r < 8; ++r) {
    int M = r + 8 * hi, M2 = 16 + M;
    float i0 = 1.f / rl[M], i1 = 1.f / rl[M2];
    size_t row0 = (size_t)b * LSEQ + mtile * 32 + M;
    size_t row1 = row0 + 16;
    ao[row0 * DMODEL + hh * DKH + n]      = f2bf_bits(O00[r] * i0);
    ao[row0 * DMODEL + hh * DKH + 16 + n] = f2bf_bits(O01[r] * i0);
    ao[row1 * DMODEL + hh * DKH + n]      = f2bf_bits(O10[r] * i1);
    ao[row1 * DMODEL + hh * DKH + 16 + n] = f2bf_bits(O11[r] * i1);
  }
}

// ---------------------------------------------------------------------------
// Output projection + residual: x = ao @ Wo + x
// ---------------------------------------------------------------------------
__global__ void k_wo(const unsigned short* __restrict__ ao,
                     const unsigned short* __restrict__ WoT,
                     float* __restrict__ xf, unsigned short* __restrict__ xbf, int layer) {
  int wid = blockIdx.x * (blockDim.x >> 5) + (threadIdx.x >> 5);
  int lane = threadIdx.x & 31;
  int mtile = wid >> 3, ntile = wid & 7;
  const unsigned short* Arow = ao + (size_t)mtile * 16 * DMODEL;
  const unsigned short* Brow = WoT + (size_t)layer * DMODEL * DMODEL + (size_t)ntile * 16 * DMODEL;
  v8f c = {};
#pragma unroll
  for (int kc = 0; kc < 4; ++kc)
    c = wmma_bf16(load_A(Arow + kc * 32, DMODEL, lane),
                  load_B(Brow + kc * 32, DMODEL, lane), c);
  int n = lane & 15, hi = lane >> 4;
#pragma unroll
  for (int r = 0; r < 8; ++r) {
    int M = r + 8 * hi;
    size_t idx = ((size_t)mtile * 16 + M) * DMODEL + ntile * 16 + n;
    float v = c[r] + xf[idx];
    xf[idx] = v;
    xbf[idx] = f2bf_bits(v);
  }
}

// ---------------------------------------------------------------------------
// FFN layer 1: f = gelu_exact(x @ W1 + b1)
// ---------------------------------------------------------------------------
__global__ void k_ffn1(const unsigned short* __restrict__ xbf,
                       const unsigned short* __restrict__ W1T,
                       const float* __restrict__ b1,
                       unsigned short* __restrict__ ffh, int layer) {
  int wid = blockIdx.x * (blockDim.x >> 5) + (threadIdx.x >> 5);
  int lane = threadIdx.x & 31;
  int mtile = wid >> 4, ntile = wid & 15;
  const unsigned short* Arow = xbf + (size_t)mtile * 16 * DMODEL;
  const unsigned short* Brow = W1T + (size_t)layer * DFF * DMODEL + (size_t)ntile * 16 * DMODEL;
  v8f c = {};
#pragma unroll
  for (int kc = 0; kc < 4; ++kc)
    c = wmma_bf16(load_A(Arow + kc * 32, DMODEL, lane),
                  load_B(Brow + kc * 32, DMODEL, lane), c);
  int n = lane & 15, hi = lane >> 4;
#pragma unroll
  for (int r = 0; r < 8; ++r) {
    int M = r + 8 * hi;
    int col = ntile * 16 + n;
    float v = c[r] + b1[layer * DFF + col];
    float g = 0.5f * v * (1.f + erff(v * 0.70710678118654752f));
    ffh[((size_t)mtile * 16 + M) * DFF + col] = f2bf_bits(g);
  }
}

// ---------------------------------------------------------------------------
// FFN layer 2 + residual: x = f @ W2 + b2 + x
// ---------------------------------------------------------------------------
__global__ void k_ffn2(const unsigned short* __restrict__ ffh,
                       const unsigned short* __restrict__ W2T,
                       const float* __restrict__ b2,
                       float* __restrict__ xf, unsigned short* __restrict__ xbf, int layer) {
  int wid = blockIdx.x * (blockDim.x >> 5) + (threadIdx.x >> 5);
  int lane = threadIdx.x & 31;
  int mtile = wid >> 3, ntile = wid & 7;
  const unsigned short* Arow = ffh + (size_t)mtile * 16 * DFF;
  const unsigned short* Brow = W2T + (size_t)layer * DMODEL * DFF + (size_t)ntile * 16 * DFF;
  v8f c = {};
#pragma unroll
  for (int kc = 0; kc < 8; ++kc)
    c = wmma_bf16(load_A(Arow + kc * 32, DFF, lane),
                  load_B(Brow + kc * 32, DFF, lane), c);
  int n = lane & 15, hi = lane >> 4;
#pragma unroll
  for (int r = 0; r < 8; ++r) {
    int M = r + 8 * hi;
    int col = ntile * 16 + n;
    size_t idx = ((size_t)mtile * 16 + M) * DMODEL + col;
    float v = c[r] + b2[layer * DMODEL + col] + xf[idx];
    xf[idx] = v;
    xbf[idx] = f2bf_bits(v);
  }
}

// ---------------------------------------------------------------------------
// Fusion attention logits: att = tanh(x@Wf + bf) @ uf + maskFMIN   (block=(b,l))
// ---------------------------------------------------------------------------
__global__ void k_fatt(const float* __restrict__ xf, const float* __restrict__ Wf,
                       const float* __restrict__ bfv, const float* __restrict__ uf,
                       const unsigned char* __restrict__ mask, float* __restrict__ att) {
  int bl = blockIdx.x, t = threadIdx.x;   // 128 threads
  __shared__ float xr[DMODEL];
  __shared__ float red[DMODEL];
  xr[t] = xf[(size_t)bl * DMODEL + t];
  __syncthreads();
  float a = bfv[t];
  for (int d2 = 0; d2 < DMODEL; ++d2) a += xr[d2] * Wf[d2 * DMODEL + t];
  red[t] = tanhf(a) * uf[t];
  __syncthreads();
  for (int s2 = 64; s2 > 0; s2 >>= 1) {
    if (t < s2) red[t] += red[t + s2];
    __syncthreads();
  }
  if (t == 0) att[bl] = red[0] + (mask[bl] ? 0.f : FMINV);
}

// ---------------------------------------------------------------------------
// Pooling softmax over L + head GEMV   (block=batch row, 256 threads)
// ---------------------------------------------------------------------------
__global__ void k_head(const float* __restrict__ xf, const float* __restrict__ att,
                       const float* __restrict__ demo, const float* __restrict__ Wh,
                       const float* __restrict__ bh, float* __restrict__ out) {
  int b = blockIdx.x, t = threadIdx.x;
  __shared__ float red[256];
  __shared__ float stat[2];
  __shared__ float emb[2 * DMODEL];
  float mv = -INFINITY;
  for (int l = t; l < LSEQ; l += 256) mv = fmaxf(mv, att[(size_t)b * LSEQ + l]);
  red[t] = mv; __syncthreads();
  for (int s = 128; s > 0; s >>= 1) { if (t < s) red[t] = fmaxf(red[t], red[t + s]); __syncthreads(); }
  if (t == 0) stat[0] = red[0];
  __syncthreads();
  float sv = 0.f;
  for (int l = t; l < LSEQ; l += 256) sv += expf(att[(size_t)b * LSEQ + l] - stat[0]);
  red[t] = sv; __syncthreads();
  for (int s = 128; s > 0; s >>= 1) { if (t < s) red[t] += red[t + s]; __syncthreads(); }
  if (t == 0) stat[1] = red[0];
  __syncthreads();
  if (t < DMODEL) {
    float acc = 0.f;
    for (int l = 0; l < LSEQ; ++l) {
      float w = expf(att[(size_t)b * LSEQ + l] - stat[0]);
      acc += xf[((size_t)b * LSEQ + l) * DMODEL + t] * w;
    }
    emb[t] = acc / stat[1];
    emb[DMODEL + t] = demo[b * DMODEL + t];
  }
  __syncthreads();
  if (t < FOUT) {
    float a = bh[t];
    for (int j = 0; j < 2 * DMODEL; ++j) a += emb[j] * Wh[j * FOUT + t];
    out[b * FOUT + t] = a;
  }
}

// ---------------------------------------------------------------------------
extern "C" void kernel_launch(void* const* d_in, const int* in_sizes, int n_in,
                              void* d_out, int out_size, void* d_ws, size_t ws_size,
                              hipStream_t stream) {
  (void)in_sizes; (void)n_in; (void)out_size; (void)ws_size;
  const float* values   = (const float*)d_in[0];
  const float* times    = (const float*)d_in[1];
  const int*   vars     = (const int*)d_in[2];
  const unsigned char* mask = (const unsigned char*)d_in[3];   // bool array
  const float* demog    = (const float*)d_in[4];
  const float* W1t = (const float*)d_in[5];
  const float* b1t = (const float*)d_in[6];
  const float* W2t = (const float*)d_in[7];
  const float* W1v = (const float*)d_in[8];
  const float* b1v = (const float*)d_in[9];
  const float* W2v = (const float*)d_in[10];
  const float* var_table = (const float*)d_in[11];
  const float* Wq = (const float*)d_in[12];
  const float* Wk = (const float*)d_in[13];
  const float* Wv = (const float*)d_in[14];
  const float* Wo = (const float*)d_in[15];
  const float* W1 = (const float*)d_in[16];
  const float* b1 = (const float*)d_in[17];
  const float* W2 = (const float*)d_in[18];
  const float* b2 = (const float*)d_in[19];
  const float* Wf = (const float*)d_in[20];
  const float* bfv = (const float*)d_in[21];
  const float* uf = (const float*)d_in[22];
  const float* Wd1 = (const float*)d_in[23];
  const float* bd1 = (const float*)d_in[24];
  const float* Wd2 = (const float*)d_in[25];
  const float* bd2 = (const float*)d_in[26];
  const float* Wh = (const float*)d_in[27];
  const float* bhp = (const float*)d_in[28];
  float* out = (float*)d_out;

  // Workspace layout (bytes). ffh overlaps q/k (dead by the time FFN runs).
  const size_t MB = 1u << 20;
  char* ws = (char*)d_ws;
  float*          xf  = (float*)ws;                              // 16 MB
  unsigned short* xbf = (unsigned short*)(ws + 16 * MB);         //  8 MB
  unsigned short* qb  = (unsigned short*)(ws + 24 * MB);         //  8 MB
  unsigned short* kb  = (unsigned short*)(ws + 32 * MB);         //  8 MB
  unsigned short* vT  = (unsigned short*)(ws + 40 * MB);         //  8 MB
  unsigned short* ao  = (unsigned short*)(ws + 48 * MB);         //  8 MB
  unsigned short* ffh = (unsigned short*)(ws + 24 * MB);         // 16 MB (overlap)
  unsigned short* wqT = (unsigned short*)(ws + 56 * MB);
  unsigned short* wkT = wqT + NLAYER * DMODEL * DMODEL;
  unsigned short* wvT = wkT + NLAYER * DMODEL * DMODEL;
  unsigned short* woT = wvT + NLAYER * DMODEL * DMODEL;
  unsigned short* w1T = woT + NLAYER * DMODEL * DMODEL;
  unsigned short* w2T = w1T + NLAYER * DMODEL * DFF;
  float* demo = (float*)(ws + 57 * MB + 512 * 1024);
  float* att  = demo + BSZ * DMODEL;

  // 1) weight conversion (transposed bf16 for WMMA B-operand)
  k_convT<<<(NLAYER * NHEAD * DMODEL * DKH + 255) / 256, 256, 0, stream>>>(Wq, wqT, NLAYER * NHEAD, DMODEL, DKH);
  k_convT<<<(NLAYER * NHEAD * DMODEL * DKH + 255) / 256, 256, 0, stream>>>(Wk, wkT, NLAYER * NHEAD, DMODEL, DKH);
  k_convT<<<(NLAYER * NHEAD * DMODEL * DKH + 255) / 256, 256, 0, stream>>>(Wv, wvT, NLAYER * NHEAD, DMODEL, DKH);
  k_convT<<<(NLAYER * DMODEL * DMODEL + 255) / 256, 256, 0, stream>>>(Wo, woT, NLAYER, DMODEL, DMODEL);
  k_convT<<<(NLAYER * DMODEL * DFF + 255) / 256, 256, 0, stream>>>(W1, w1T, NLAYER, DMODEL, DFF);
  k_convT<<<(NLAYER * DFF * DMODEL + 255) / 256, 256, 0, stream>>>(W2, w2T, NLAYER, DFF, DMODEL);

  // 2) embeddings + demographics
  k_embed<<<BSZ * LSEQ, DMODEL, 0, stream>>>(values, times, vars, W1t, b1t, W2t,
                                             W1v, b1v, W2v, var_table, xf, xbf);
  k_demo<<<BSZ, 2 * DMODEL, 0, stream>>>(demog, Wd1, bd1, Wd2, bd2, demo);

  // 3) transformer stack
  const int MT = (BSZ * LSEQ) / 16;   // 2048 row tiles
  for (int layer = 0; layer < NLAYER; ++layer) {
    k_qkv<<<(MT * 24) / 8, 256, 0, stream>>>(xbf, wqT, wkT, wvT, qb, kb, vT, layer);
    k_attn<<<BSZ * NHEAD * (LSEQ / 32), 32, 0, stream>>>(qb, kb, vT, mask, ao);
    k_wo<<<(MT * 8) / 8, 256, 0, stream>>>(ao, woT, xf, xbf, layer);
    k_ffn1<<<(MT * 16) / 8, 256, 0, stream>>>(xbf, w1T, b1, ffh, layer);
    k_ffn2<<<(MT * 8) / 8, 256, 0, stream>>>(ffh, w2T, b2, xf, xbf, layer);
  }

  // 4) fusion pooling + head
  k_fatt<<<BSZ * LSEQ, DMODEL, 0, stream>>>(xf, Wf, bfv, uf, mask, att);
  k_head<<<BSZ, 256, 0, stream>>>(xf, att, demo, Wh, bhp, out);
}
